// PointSemantic_dense_9096740733635
// MI455X (gfx1250) — compile-verified, matrix operand order checked by reading, not run on hardware
//
#include <hip/hip_runtime.h>
#include <math.h>

// ---------------------------------------------------------------------------
// CDNA5 (gfx1250) implementation of the PointNet++ dense-segmentation net.
// GEMMs (all cbr layers, FP layers, head) use v_wmma_f32_16x16x32_bf16 with
// f32 accumulation; dense-block concat is realized by writing each layer's
// output channels directly into a strided activation buffer.
// ---------------------------------------------------------------------------

typedef __attribute__((ext_vector_type(16))) __bf16          v16bf;
typedef __attribute__((ext_vector_type(8)))  float           v8f;
typedef __attribute__((ext_vector_type(8)))  unsigned short  ushort8;

#define KB 4          // batch
#define KN 8192       // points
#define KNS 32        // nsample
#define KNC 13        // classes

__device__ __forceinline__ unsigned short f2bfbits(float f) {
  unsigned int u = __builtin_bit_cast(unsigned int, f);
  unsigned int r = u + 0x7FFFu + ((u >> 16) & 1u);   // RTNE
  return (unsigned short)(r >> 16);
}

// ------------------------- farthest point sampling -------------------------
// One workgroup per batch; dist array lives in LDS (N<=8192 -> 32KB of the
// 320KB WGP LDS). 1024 dependent iterations: min-update + tree argmax.
__global__ __launch_bounds__(256) void fps_kernel(const float* __restrict__ xyz,
                                                  int N, int S, int* __restrict__ out) {
  __shared__ float s_dist[KN];
  __shared__ float rv[256];
  __shared__ int   ri[256];
  __shared__ int   s_far;
  const int b = blockIdx.x, t = threadIdx.x, T = blockDim.x;
  const float* x = xyz + (size_t)b * N * 3;
  for (int i = t; i < N; i += T) s_dist[i] = 1e10f;
  if (t == 0) s_far = 0;
  __syncthreads();
  for (int it = 0; it < S; ++it) {
    int far = s_far;
    if (t == 0) out[b * S + it] = far;
    float fx = x[far * 3 + 0], fy = x[far * 3 + 1], fz = x[far * 3 + 2];
    float best = -1e30f; int bidx = 0;
    for (int i = t; i < N; i += T) {
      float dx = x[i * 3 + 0] - fx, dy = x[i * 3 + 1] - fy, dz = x[i * 3 + 2] - fz;
      float d = dx * dx + dy * dy + dz * dz;
      float nd = fminf(s_dist[i], d);
      s_dist[i] = nd;
      if (nd > best) { best = nd; bidx = i; }           // ascending i -> first max
    }
    rv[t] = best; ri[t] = bidx;
    __syncthreads();
    for (int s2 = T / 2; s2 > 0; s2 >>= 1) {
      if (t < s2) {
        if (rv[t + s2] > rv[t] || (rv[t + s2] == rv[t] && ri[t + s2] < ri[t])) {
          rv[t] = rv[t + s2]; ri[t] = ri[t + s2];
        }
      }
      __syncthreads();
    }
    if (t == 0) s_far = ri[0];
    __syncthreads();
  }
}

__global__ void gather_xyz_kernel(const float* __restrict__ xyz, const int* __restrict__ fidx,
                                  int N, int S, float* __restrict__ out) {
  int i = blockIdx.x * blockDim.x + threadIdx.x;
  if (i >= KB * S) return;
  int b = i / S;
  int p = fidx[i];
  const float* s = xyz + (size_t)(b * N + p) * 3;
  out[i * 3 + 0] = s[0]; out[i * 3 + 1] = s[1]; out[i * 3 + 2] = s[2];
}

// Ball query == first `nsample` in-range indices in ascending order
// (exactly what the reference's where+sort+slice computes).
__global__ void ball_query_kernel(const float* __restrict__ xyz, const float* __restrict__ nxyz,
                                  int N, int S, float r2, int* __restrict__ gidx) {
  int i = blockIdx.x * blockDim.x + threadIdx.x;
  if (i >= KB * S) return;
  int b = i / S;
  const float* q = nxyz + (size_t)i * 3;
  const float* x = xyz + (size_t)b * N * 3;
  int* o = gidx + (size_t)i * KNS;
  int cnt = 0, first = N - 1;
  for (int j = 0; j < N && cnt < KNS; ++j) {
    float dx = x[j * 3 + 0] - q[0], dy = x[j * 3 + 1] - q[1], dz = x[j * 3 + 2] - q[2];
    if (dx * dx + dy * dy + dz * dz <= r2) {
      if (cnt == 0) first = j;
      o[cnt++] = j;
    }
  }
  for (; cnt < KNS; ++cnt) o[cnt] = first;
}

// Grouped feature build: row = (b,s,j) -> [xyz[p]-new_xyz | feats[p]]
__global__ void group_kernel(const float* __restrict__ xyz, const float* __restrict__ nxyz,
                             const float* __restrict__ feats, int ldf,
                             const int* __restrict__ gidx, int N, int S, int Cf,
                             float* __restrict__ out, int ldo) {
  int row = blockIdx.x;              // b*S*NS + s*NS + j
  int g = row / KNS;
  int b = g / S;
  int p = gidx[row];
  const float* src = xyz + (size_t)(b * N + p) * 3;
  const float* q = nxyz + (size_t)g * 3;
  float* o = out + (size_t)row * ldo;
  for (int c = threadIdx.x; c < 3 + Cf; c += blockDim.x)
    o[c] = (c < 3) ? (src[c] - q[c]) : feats[(size_t)(b * N + p) * ldf + (c - 3)];
}

// ------------------------------ WMMA GEMM ----------------------------------
// Y[M,N] = act((X[M,K] @ W[K,N] + preb) * g + bias), bf16 inputs, f32 accum.
// Block = 4 waves; block tile 64x32. Each wave owns 16 rows and BOTH 16-col
// groups: one A fragment feeds two WMMAs (two accumulators) per K-step.
// LDS pitch 40 ushorts (80B) => each lane's 16 fragment elements are two
// 16B-aligned contiguous runs -> 2x ds_load_b128 per fragment. B is staged
// transposed so its fragment reads have the same shape.
// Staging uses clamped addresses + value predication (no exec-mask branches):
// the global load is always in-bounds, only the stored value is selected.
#define APITCH 40
__global__ __launch_bounds__(128)
void gemm_cbr_wmma_kernel(const float* __restrict__ X, int lda,
                          const float* __restrict__ W,
                          const float* __restrict__ preb,
                          const float* __restrict__ gsc,
                          const float* __restrict__ bias,
                          float* __restrict__ Y, int ldc,
                          int M, int N, int K, int relu) {
  __shared__ __align__(16) unsigned short As[64][APITCH];    // [row][k]
  __shared__ __align__(16) unsigned short BsT[32][APITCH];   // [n][k] (transposed)
  const int tid = threadIdx.x;
  const int wave = tid >> 5;
  const int lane = tid & 31;
  const int hlf = lane >> 4;
  const int mn = lane & 15;
  const int m0 = blockIdx.x * 64;
  const int n0 = blockIdx.y * 32;
  v8f acc0 = {0.f, 0.f, 0.f, 0.f, 0.f, 0.f, 0.f, 0.f};
  v8f acc1 = {0.f, 0.f, 0.f, 0.f, 0.f, 0.f, 0.f, 0.f};

  for (int k0 = 0; k0 < K; k0 += 32) {
    // stage A: 64 rows x 32 K (coalesced over k); clamped addr, predicated value
    for (int i = tid; i < 64 * 32; i += 128) {
      int r = i >> 5, k = i & 31;
      int gr = m0 + r, gk = k0 + k;
      int cr = (gr < M) ? gr : (M - 1);
      int ck = (gk < K) ? gk : (K - 1);
      float v = X[(size_t)cr * lda + ck];
      As[r][k] = (gr < M && gk < K) ? f2bfbits(v) : (unsigned short)0;
    }
    // stage B transposed: 32 cols x 32 K (coalesced over n)
    for (int i = tid; i < 32 * 32; i += 128) {
      int k = i >> 5, n = i & 31;
      int gk = k0 + k, gn = n0 + n;
      int ck = (gk < K) ? gk : (K - 1);
      int cn = (gn < N) ? gn : (N - 1);
      float v = W[(size_t)ck * N + cn];
      BsT[n][k] = (gk < K && gn < N) ? f2bfbits(v) : (unsigned short)0;
    }
    __syncthreads();
    // 16-bit A 16x32 layout: lane<16 -> K 0..7,16..23 ; lane>=16 -> K 8..15,24..31
    // => elements [0..7] = run at k=hlf*8, elements [8..15] = run at k=16+hlf*8.
    ushort8 ra0 = *(const ushort8*)&As[wave * 16 + mn][hlf * 8];
    ushort8 ra1 = *(const ushort8*)&As[wave * 16 + mn][16 + hlf * 8];
    ushort8 rb0 = *(const ushort8*)&BsT[mn][hlf * 8];
    ushort8 rb1 = *(const ushort8*)&BsT[mn][16 + hlf * 8];
    ushort8 rc0 = *(const ushort8*)&BsT[16 + mn][hlf * 8];
    ushort8 rc1 = *(const ushort8*)&BsT[16 + mn][16 + hlf * 8];
    v16bf a = __builtin_bit_cast(v16bf, __builtin_shufflevector(
        ra0, ra1, 0, 1, 2, 3, 4, 5, 6, 7, 8, 9, 10, 11, 12, 13, 14, 15));
    v16bf b0 = __builtin_bit_cast(v16bf, __builtin_shufflevector(
        rb0, rb1, 0, 1, 2, 3, 4, 5, 6, 7, 8, 9, 10, 11, 12, 13, 14, 15));
    v16bf b1 = __builtin_bit_cast(v16bf, __builtin_shufflevector(
        rc0, rc1, 0, 1, 2, 3, 4, 5, 6, 7, 8, 9, 10, 11, 12, 13, 14, 15));
    acc0 = __builtin_amdgcn_wmma_f32_16x16x32_bf16(false, a, false, b0,
                                                   (short)0, acc0, false, false);
    acc1 = __builtin_amdgcn_wmma_f32_16x16x32_bf16(false, a, false, b1,
                                                   (short)0, acc1, false, false);
    __syncthreads();
  }

#pragma unroll
  for (int grp = 0; grp < 2; ++grp) {
    int gn = n0 + grp * 16 + mn;
    if (gn < N) {
      const v8f& acc = grp ? acc1 : acc0;
      float pb = preb ? preb[gn] : 0.0f;
      float gv = gsc ? gsc[gn] : 1.0f;
      float bv = bias ? bias[gn] : 0.0f;
#pragma unroll
      for (int r = 0; r < 8; ++r) {
        int gm = m0 + wave * 16 + hlf * 8 + r;   // C/D layout: VGPR r -> M = r + 8*half
        if (gm < M) {
          float y = (acc[r] + pb) * gv + bv;
          if (relu) y = fmaxf(y, 0.0f);
          Y[(size_t)gm * ldc + gn] = y;
        }
      }
    }
  }
}

__global__ void maxpool_kernel(const float* __restrict__ x, int ld, int C,
                               float* __restrict__ out, int ldo) {
  int g = blockIdx.x;                        // group b*S+s
  const float* base = x + (size_t)g * KNS * ld;
  for (int c = threadIdx.x; c < C; c += blockDim.x) {
    float m = base[c];
    for (int j = 1; j < KNS; ++j) m = fmaxf(m, base[(size_t)j * ld + c]);
    out[(size_t)g * ldo + c] = m;
  }
}

__global__ void knn3_kernel(const float* __restrict__ xyz1, const float* __restrict__ xyz2,
                            int N1, int N2, int* __restrict__ idx3, float* __restrict__ w3) {
  int i = blockIdx.x * blockDim.x + threadIdx.x;
  if (i >= KB * N1) return;
  int b = i / N1;
  const float* q = xyz1 + (size_t)i * 3;
  const float* x = xyz2 + (size_t)b * N2 * 3;
  float d0 = 1e30f, d1 = 1e30f, d2 = 1e30f;
  int i0 = 0, i1 = 0, i2 = 0;
  for (int j = 0; j < N2; ++j) {
    float dx = x[j * 3 + 0] - q[0], dy = x[j * 3 + 1] - q[1], dz = x[j * 3 + 2] - q[2];
    float d = dx * dx + dy * dy + dz * dz;
    if (d < d0)      { d2 = d1; i2 = i1; d1 = d0; i1 = i0; d0 = d; i0 = j; }
    else if (d < d1) { d2 = d1; i2 = i1; d1 = d; i1 = j; }
    else if (d < d2) { d2 = d; i2 = j; }
  }
  float a = 1.0f / (fmaxf(d0, 0.0f) + 1e-8f);
  float c = 1.0f / (fmaxf(d1, 0.0f) + 1e-8f);
  float e = 1.0f / (fmaxf(d2, 0.0f) + 1e-8f);
  float s = a + c + e;
  idx3[i * 3 + 0] = i0; idx3[i * 3 + 1] = i1; idx3[i * 3 + 2] = i2;
  w3[i * 3 + 0] = a / s; w3[i * 3 + 1] = c / s; w3[i * 3 + 2] = e / s;
}

__global__ void interp_kernel(const float* __restrict__ pts2, int ld2, int N2,
                              const int* __restrict__ idx3, const float* __restrict__ w3,
                              int N1, int C, float* __restrict__ out, int ldo, int coff) {
  int row = blockIdx.x;                      // b*N1 + i
  int b = row / N1;
  const int* id = idx3 + (size_t)row * 3;
  const float* w = w3 + (size_t)row * 3;
  const float* p0 = pts2 + (size_t)(b * N2 + id[0]) * ld2;
  const float* p1 = pts2 + (size_t)(b * N2 + id[1]) * ld2;
  const float* p2 = pts2 + (size_t)(b * N2 + id[2]) * ld2;
  for (int c = threadIdx.x; c < C; c += blockDim.x)
    out[(size_t)row * ldo + coff + c] = w[0] * p0[c] + w[1] * p1[c] + w[2] * p2[c];
}

__global__ void copycols_kernel(const float* __restrict__ src, int lds_, int C,
                                float* __restrict__ dst, int ldd, int coff) {
  int row = blockIdx.x;
  for (int c = threadIdx.x; c < C; c += blockDim.x)
    dst[(size_t)row * ldd + coff + c] = src[(size_t)row * lds_ + c];
}

__global__ void logsoftmax_kernel(float* __restrict__ x, int C, int rows) {
  int row = blockIdx.x * blockDim.x + threadIdx.x;
  if (row >= rows) return;
  float* p = x + (size_t)row * C;
  float m = p[0];
  for (int c = 1; c < C; ++c) m = fmaxf(m, p[c]);
  float s = 0.0f;
  for (int c = 0; c < C; ++c) s += __expf(p[c] - m);
  float ls = __logf(s);
  for (int c = 0; c < C; ++c) p[c] = p[c] - m - ls;
}

// ------------------------------- host side ---------------------------------
static inline void gemm(hipStream_t st, const float* X, int lda, const float* W,
                        const float* preb, const float* g, const float* b,
                        float* Y, int ldc, int M, int N, int K, int relu) {
  dim3 grid((M + 63) / 64, (N + 31) / 32);
  gemm_cbr_wmma_kernel<<<grid, 128, 0, st>>>(X, lda, W, preb, g, b, Y, ldc, M, N, K, relu);
}

struct CW { const float* w; const float* g; const float* b; };

extern "C" void kernel_launch(void* const* d_in, const int* in_sizes, int n_in,
                              void* d_out, int out_size, void* d_ws, size_t ws_size,
                              hipStream_t stream) {
  (void)n_in; (void)out_size; (void)ws_size;
  const int L[4]  = {6, 12, 18, 24};
  const int np[4] = {1024, 256, 64, 16};
  const float rad[4] = {0.5f, 1.0f, 2.0f, 4.0f};

  // ---- locate weights: insertion order (pc first) vs jax-sorted order ----
  CW sa_conv0[4], sa_dense[4][24], sa_trans[4], fp_l[4][2];
  const float *h_w1, *h_b1, *h_g1, *h_beta1, *h_w2, *h_b2, *pc;
  bool insertion = (in_sizes[0] == KB * KN * 3);
  if (insertion) {
    pc = (const float*)d_in[0];
    int idx = 1;
    auto take = [&](CW& c) {
      c.w = (const float*)d_in[idx]; c.g = (const float*)d_in[idx + 1];
      c.b = (const float*)d_in[idx + 2]; idx += 3;
    };
    for (int s = 0; s < 4; ++s) {
      take(sa_conv0[s]);
      for (int j = 0; j < L[s]; ++j) take(sa_dense[s][j]);
      take(sa_trans[s]);
    }
    for (int f = 0; f < 4; ++f) { take(fp_l[f][0]); take(fp_l[f][1]); }
    h_w1 = (const float*)d_in[idx]; h_b1 = (const float*)d_in[idx + 1];
    h_g1 = (const float*)d_in[idx + 2]; h_beta1 = (const float*)d_in[idx + 3];
    h_w2 = (const float*)d_in[idx + 4]; h_b2 = (const float*)d_in[idx + 5];
  } else {
    int idx = 0;
    auto take = [&](CW& c) {   // sorted conv keys: b < g < w
      c.b = (const float*)d_in[idx]; c.g = (const float*)d_in[idx + 1];
      c.w = (const float*)d_in[idx + 2]; idx += 3;
    };
    for (int f = 0; f < 4; ++f) { take(fp_l[f][0]); take(fp_l[f][1]); }
    h_b1 = (const float*)d_in[idx]; h_b2 = (const float*)d_in[idx + 1];
    h_beta1 = (const float*)d_in[idx + 2]; h_g1 = (const float*)d_in[idx + 3];
    h_w1 = (const float*)d_in[idx + 4]; h_w2 = (const float*)d_in[idx + 5];
    idx += 6;
    for (int s = 0; s < 4; ++s) {
      take(sa_conv0[s]);
      for (int j = 0; j < L[s]; ++j) take(sa_dense[s][j]);
      take(sa_trans[s]);
    }
    pc = (const float*)d_in[idx];
  }

  // ---- channel bookkeeping ----
  int cin0[4], cfull[4], coutc[4];
  int cf = 0;
  for (int s = 0; s < 4; ++s) {
    cin0[s] = cf + 3;
    int c = 2 * cin0[s] + 32 * L[s];
    cfull[s] = c;
    coutc[s] = c / 2;
    cf = coutc[s];
  }
  // cin0 = {3,102,297,588}; cfull = {198,588,1170,1944}; coutc = {99,294,585,972}

  // ---- workspace bump allocator ----
  char* wsp = (char*)d_ws;
  size_t off = 0;
  auto allocF = [&](size_t n) { float* p = (float*)(wsp + off); off = (off + n * 4 + 255) & ~(size_t)255; return p; };
  auto allocI = [&](size_t n) { int* p = (int*)(wsp + off); off = (off + n * 4 + 255) & ~(size_t)255; return p; };

  // persistent
  float* xyz_p[5]; xyz_p[0] = nullptr;     // pc is input
  for (int s = 0; s < 4; ++s) xyz_p[s + 1] = allocF((size_t)KB * np[s] * 3);
  float* fbuf[4];
  for (int s = 0; s < 4; ++s) fbuf[s] = allocF((size_t)KB * np[s] * coutc[s]);
  int* fps_idx = allocI((size_t)KB * 1024);
  int* gidx    = allocI((size_t)KB * 1024 * KNS);
  int* knn_idx = allocI((size_t)KB * KN * 3);
  float* knn_w = allocF((size_t)KB * KN * 3);
  float* l3b = allocF((size_t)KB * 64 * 585);
  float* l2b = allocF((size_t)KB * 256 * 585);
  float* l1b = allocF((size_t)KB * 1024 * 294);
  float* l0b = allocF((size_t)KB * KN * 128);
  float* hdy = allocF((size_t)KB * KN * 128);
  size_t stage_base = off;

  // ---- SA stages ----
  const float* cur_xyz = pc; int curN = KN;
  const float* cur_f = nullptr; int curC = 0, curLD = 0;
  for (int s = 0; s < 4; ++s) {
    off = stage_base;
    int S = np[s];
    float r2 = rad[s] * rad[s];
    int rows = KB * S * KNS;
    int cin = curC + 3;
    float* G   = allocF((size_t)rows * cin);
    float* ACT = allocF((size_t)rows * cfull[s]);
    float* T   = allocF((size_t)rows * coutc[s]);

    fps_kernel<<<KB, 256, 0, stream>>>(cur_xyz, curN, S, fps_idx);
    gather_xyz_kernel<<<(KB * S + 127) / 128, 128, 0, stream>>>(cur_xyz, fps_idx, curN, S, xyz_p[s + 1]);
    ball_query_kernel<<<(KB * S + 63) / 64, 64, 0, stream>>>(cur_xyz, xyz_p[s + 1], curN, S, r2, gidx);
    group_kernel<<<rows, 64, 0, stream>>>(cur_xyz, xyz_p[s + 1], cur_f, curLD, gidx, curN, S, curC, G, cin);

    gemm(stream, G, cin, sa_conv0[s].w, nullptr, sa_conv0[s].g, sa_conv0[s].b,
         ACT, cfull[s], rows, 2 * cin, cin, 1);
    int ccur = 2 * cin;
    for (int j = 0; j < L[s]; ++j) {                 // dense concat = strided write
      gemm(stream, ACT, cfull[s], sa_dense[s][j].w, nullptr, sa_dense[s][j].g, sa_dense[s][j].b,
           ACT + ccur, cfull[s], rows, 32, ccur, 1);
      ccur += 32;
    }
    gemm(stream, ACT, cfull[s], sa_trans[s].w, nullptr, sa_trans[s].g, sa_trans[s].b,
         T, coutc[s], rows, coutc[s], ccur, 1);
    maxpool_kernel<<<KB * S, 128, 0, stream>>>(T, coutc[s], coutc[s], fbuf[s], coutc[s]);

    cur_xyz = xyz_p[s + 1]; curN = S;
    cur_f = fbuf[s]; curC = coutc[s]; curLD = coutc[s];
  }

  // ---- FP stages ----
  auto fp_stage = [&](const float* x1, int N1, const float* x2, int N2,
                      const float* p1, int C1, int ld1,
                      const float* p2, int C2, int ld2,
                      CW la, CW lb, int mid, int outC, float* outB) {
    off = stage_base;
    int rows = KB * N1;
    int cin = C1 + C2;
    float* IN  = allocF((size_t)rows * cin);
    float* MID = allocF((size_t)rows * mid);
    knn3_kernel<<<(rows + 63) / 64, 64, 0, stream>>>(x1, x2, N1, N2, knn_idx, knn_w);
    if (p1) copycols_kernel<<<rows, 128, 0, stream>>>(p1, ld1, C1, IN, cin, 0);
    interp_kernel<<<rows, 128, 0, stream>>>(p2, ld2, N2, knn_idx, knn_w, N1, C2, IN, cin, C1);
    gemm(stream, IN, cin, la.w, nullptr, la.g, la.b, MID, mid, rows, mid, cin, 1);
    gemm(stream, MID, mid, lb.w, nullptr, lb.g, lb.b, outB, outC, rows, outC, mid, 1);
  };
  fp_stage(xyz_p[3], 64,   xyz_p[4], 16,  fbuf[2], coutc[2], coutc[2], fbuf[3], coutc[3], coutc[3],
           fp_l[0][0], fp_l[0][1], coutc[2], coutc[2], l3b);          // 1557 -> 585 -> 585
  fp_stage(xyz_p[2], 256,  xyz_p[3], 64,  fbuf[1], coutc[1], coutc[1], l3b, coutc[2], coutc[2],
           fp_l[1][0], fp_l[1][1], coutc[2], coutc[2], l2b);          // 879 -> 585 -> 585
  fp_stage(xyz_p[1], 1024, xyz_p[2], 256, fbuf[0], coutc[0], coutc[0], l2b, coutc[2], coutc[2],
           fp_l[2][0], fp_l[2][1], coutc[2], coutc[1], l1b);          // 684 -> 585 -> 294
  fp_stage(pc, KN, xyz_p[1], 1024, nullptr, 0, 0, l1b, coutc[1], coutc[1],
           fp_l[3][0], fp_l[3][1], coutc[1], 128, l0b);               // 294 -> 294 -> 128

  // ---- head: relu((l0 @ w1 + b1)*g1 + beta1) @ w2 + b2, log_softmax ----
  gemm(stream, l0b, 128, h_w1, h_b1, h_g1, h_beta1, hdy, 128, KB * KN, 128, 128, 1);
  gemm(stream, hdy, 128, h_w2, nullptr, nullptr, h_b2, (float*)d_out, KNC, KB * KN, KNC, 128, 0);
  logsoftmax_kernel<<<(KB * KN + 127) / 128, 128, 0, stream>>>((float*)d_out, KNC, KB * KN);
}
